// MolGINE_53446573031960
// MI455X (gfx1250) — matrix operand order repaired and checked by reading, not compiled, not run on hardware
//
#include <hip/hip_runtime.h>
#include <hip/hip_bf16.h>

// ---------------------------------------------------------------------------
// MolGINE on MI455X (gfx1250): bf16 WMMA GEMMs with pre-transposed bf16
// weight panels (converted once per launch), f32 atomic scatter pipeline.
// ---------------------------------------------------------------------------

#define HID      128
#define OUTD     256
#define NNODES   200000
#define NEDGES   400000
#define NGRAPHS  8000
#define NLAYERS  3

typedef __attribute__((ext_vector_type(16))) __bf16 v16bf;
typedef __attribute__((ext_vector_type(8)))  float  v8f;

// fp32 -> bf16 round-to-nearest-even
__device__ __forceinline__ __bf16 f2bf(float f) {
    unsigned u = __float_as_uint(f);
    u += 0x7FFFu + ((u >> 16) & 1u);
    unsigned short s = (unsigned short)(u >> 16);
    __bf16 b;
    __builtin_memcpy(&b, &s, sizeof(b));
    return b;
}

// pack two fp32 -> packed bf16x2 dword (a -> low 16, b -> high 16)
__device__ __forceinline__ unsigned pack2bf(float a, float b) {
    unsigned ua = __float_as_uint(a);
    ua += 0x7FFFu + ((ua >> 16) & 1u);
    unsigned ub = __float_as_uint(b);
    ub += 0x7FFFu + ((ub >> 16) & 1u);
    return (ua >> 16) | (ub & 0xFFFF0000u);
}

// A fragment: 16x32 bf16 tile, row-major in LDS (stride: even #elements).
// Contiguous k order: lanes 0-15 rows M=lane; VGPR p: K=2p,2p+1;
// lanes 16-31: K += 16 (matches B). Pairs contiguous -> ds_load_b32 / 2addr_b32.
__device__ __forceinline__ v16bf load_a_frag(const __bf16* As, int stride) {
    const int lane = threadIdx.x & 31;
    const int half = lane >> 4, m = lane & 15;
    const __bf16* row = As + m * stride + 16 * half;
    v16bf a;
#pragma unroll
    for (int p = 0; p < 8; ++p) {
        const int kb = 2 * p;
        a[2 * p]     = row[kb];
        a[2 * p + 1] = row[kb + 1];
    }
    return a;
}

// B fragment: 32x16 bf16, loaded directly from a pre-transposed global
// weight panel WT[N][K] (k fastest). Lane (halfb*16+n) needs
// WT[n0+n][k0 + 16*halfb .. +15] = 16 contiguous bf16 = 2 x b128 loads.
__device__ __forceinline__ v16bf load_b_frag_g(const __bf16* __restrict__ WT,
                                               int K, int n0, int k0) {
    const int lane  = threadIdx.x & 31;
    const int halfb = lane >> 4, n = lane & 15;
    const __bf16* p = WT + (size_t)(n0 + n) * K + k0 + 16 * halfb;
    uint4 lo = *(const uint4*)p;        // 16B aligned: K % 32 == 0
    uint4 hi = *(const uint4*)(p + 8);
    v16bf b;
    __builtin_memcpy(&b, &lo, 16);
    __builtin_memcpy(((char*)&b) + 16, &hi, 16);
    return b;
}

__device__ __forceinline__ v8f wmma_bf16(v16bf a, v16bf b, v8f c) {
    return __builtin_amdgcn_wmma_f32_16x16x32_bf16(false, a, false, b,
                                                   (short)0, c, false, false);
}

// Packed-embedding row offsets (cumsum of vocab sizes)
template <int NF> __device__ __forceinline__ int feat_offset(int f);
template <> __device__ __forceinline__ int feat_offset<9>(int f) {
    switch (f) {
        case 0: return 0;   case 1: return 119; case 2: return 128;
        case 3: return 139; case 4: return 151; case 5: return 161;
        case 6: return 166; case 7: return 174; default: return 176;
    }
}
template <> __device__ __forceinline__ int feat_offset<3>(int f) {
    switch (f) { case 0: return 0; case 1: return 22; default: return 28; }
}

// ---------------------------------------------------------------------------
// One-time weight prep: W[K][N] fp32 -> WT[N][K] bf16 (transpose + convert)
// ---------------------------------------------------------------------------
__global__ __launch_bounds__(256) void weight_prep_kernel(
    const float* __restrict__ W, __bf16* __restrict__ WT, int K, int N)
{
    const int i = blockIdx.x * 256 + threadIdx.x;
    if (i >= K * N) return;
    const int k = i / N, n = i - k * N;
    WT[(size_t)n * K + k] = f2bf(W[i]);
}

// ---------------------------------------------------------------------------
// Embedding gather + projection: out[16 rows] = concat(emb rows) @ W + bias
// One block = 16 rows; 8 waves each own a 16-wide N tile of the 128 outputs.
// ---------------------------------------------------------------------------
template <int NF>
__global__ __launch_bounds__(256) void embed_project_kernel(
    const int* __restrict__ codes,    // [Nrows, NF]
    const float* __restrict__ emb,    // [TOT, HID]
    const __bf16* __restrict__ WT,    // [HID, NF*HID] transposed bf16 panel
    const float* __restrict__ bias,   // [HID]
    float* __restrict__ out)          // [Nrows, HID]
{
    constexpr int KT  = NF * HID;
    constexpr int AST = KT + 4;   // stride%4==0 (b64 LDS stores), 2-dword skew
    __shared__ __attribute__((aligned(16))) __bf16 As[16 * AST];
    __shared__ int rowL[16 * NF];

    const int  tid  = threadIdx.x;
    const int  wave = tid >> 5;
    const long row0 = (long)blockIdx.x * 16;

    for (int i = tid; i < 16 * NF; i += 256)
        rowL[i] = codes[row0 * NF + i] + feat_offset<NF>(i % NF);
    __syncthreads();

    // gather embeddings: float4 global load -> packed bf16x4 -> ds_store_b64
    for (int i = tid; i < 16 * (KT / 4); i += 256) {
        const int t = i / (KT / 4), c = (i - t * (KT / 4)) * 4;
        const int f = c >> 7, j = c & 127;
        const float4 v =
            *(const float4*)(emb + (size_t)rowL[t * NF + f] * HID + j);
        uint2 pk;
        pk.x = pack2bf(v.x, v.y);
        pk.y = pack2bf(v.z, v.w);
        *(uint2*)(As + t * AST + c) = pk;
    }
    __syncthreads();

    const int lane = tid & 31, half = lane >> 4, n = lane & 15;
    const int n0   = wave * 16;
    v8f acc = {0.f, 0.f, 0.f, 0.f, 0.f, 0.f, 0.f, 0.f};
#pragma unroll 4
    for (int k0 = 0; k0 < KT; k0 += 32) {
        v16bf a = load_a_frag(As + k0, AST);
        v16bf b = load_b_frag_g(WT, KT, n0, k0);
        acc = wmma_bf16(a, b, acc);
    }

    const int   col = n0 + n;
    const float bv  = bias[col];
#pragma unroll
    for (int r = 0; r < 8; ++r)
        out[(row0 + r + 8 * half) * HID + col] = acc[r] + bv;
}

// ---------------------------------------------------------------------------
// GINE layer MLP: h = relu(relu((h+agg) @ W1 + b1) @ W2 + b2), in place.
// ---------------------------------------------------------------------------
__global__ __launch_bounds__(256) void gine_mlp_kernel(
    float* __restrict__ h, const float* __restrict__ agg,
    const __bf16* __restrict__ W1T, const float* __restrict__ b1,
    const __bf16* __restrict__ W2T, const float* __restrict__ b2)
{
    constexpr int ZST = HID + 4;
    __shared__ __attribute__((aligned(16))) __bf16 Zs[16 * ZST];
    __shared__ __attribute__((aligned(16))) __bf16 Ts[16 * ZST];

    const int  tid  = threadIdx.x;
    const int  wave = tid >> 5;
    const long row0 = (long)blockIdx.x * 16;

    for (int i = tid; i < 16 * (HID / 4); i += 256) {
        const int t = i >> 5, c = (i & 31) * 4;
        const long gi = (row0 + t) * HID + c;
        const float4 hv = *(const float4*)(h + gi);
        const float4 av = *(const float4*)(agg + gi);
        uint2 pk;
        pk.x = pack2bf(hv.x + av.x, hv.y + av.y);
        pk.y = pack2bf(hv.z + av.z, hv.w + av.w);
        *(uint2*)(Zs + t * ZST + c) = pk;
    }
    __syncthreads();

    const int lane = tid & 31, half = lane >> 4, n = lane & 15;
    const int n0 = wave * 16, col = n0 + n;

    v8f acc = {0.f, 0.f, 0.f, 0.f, 0.f, 0.f, 0.f, 0.f};
#pragma unroll
    for (int k0 = 0; k0 < HID; k0 += 32)
        acc = wmma_bf16(load_a_frag(Zs + k0, ZST),
                        load_b_frag_g(W1T, HID, n0, k0), acc);

    const float bv1 = b1[col];
#pragma unroll
    for (int r = 0; r < 8; ++r) {
        const float v = acc[r] + bv1;
        Ts[(r + 8 * half) * ZST + col] = f2bf(v > 0.f ? v : 0.f);
    }
    __syncthreads();

    v8f acc2 = {0.f, 0.f, 0.f, 0.f, 0.f, 0.f, 0.f, 0.f};
#pragma unroll
    for (int k0 = 0; k0 < HID; k0 += 32)
        acc2 = wmma_bf16(load_a_frag(Ts + k0, ZST),
                         load_b_frag_g(W2T, HID, n0, k0), acc2);

    const float bv2 = b2[col];
#pragma unroll
    for (int r = 0; r < 8; ++r) {
        const float v = acc2[r] + bv2;
        h[(row0 + r + 8 * half) * HID + col] = v > 0.f ? v : 0.f;
    }
}

// ---------------------------------------------------------------------------
// Message pass: agg[dst] += relu(h[src] + ef[e]); 8 edges/block, float4 wide.
// ---------------------------------------------------------------------------
__global__ __launch_bounds__(256) void edge_message_kernel(
    const float* __restrict__ h, const float* __restrict__ ef,
    const int* __restrict__ src, const int* __restrict__ dst,
    float* __restrict__ agg)
{
    const int e = blockIdx.x * 8 + (threadIdx.x >> 5);
    const int c = (threadIdx.x & 31) * 4;
    const int s = src[e], d = dst[e];
    const float4 hv = *(const float4*)(h + (long)s * HID + c);
    const float4 ev = *(const float4*)(ef + (long)e * HID + c);
    float* ap = agg + (long)d * HID + c;
    atomicAdd(ap + 0, fmaxf(hv.x + ev.x, 0.f));
    atomicAdd(ap + 1, fmaxf(hv.y + ev.y, 0.f));
    atomicAdd(ap + 2, fmaxf(hv.z + ev.z, 0.f));
    atomicAdd(ap + 3, fmaxf(hv.w + ev.w, 0.f));
}

// global_add_pool: g[batch[n]] += h[n]; 8 nodes/block, float4 wide.
__global__ __launch_bounds__(256) void pool_kernel(
    const float* __restrict__ h, const int* __restrict__ batch,
    float* __restrict__ g)
{
    const int node = blockIdx.x * 8 + (threadIdx.x >> 5);
    const int c    = (threadIdx.x & 31) * 4;
    const int b    = batch[node];
    const float4 hv = *(const float4*)(h + (long)node * HID + c);
    float* gp = g + (long)b * HID + c;
    atomicAdd(gp + 0, hv.x);
    atomicAdd(gp + 1, hv.y);
    atomicAdd(gp + 2, hv.z);
    atomicAdd(gp + 3, hv.w);
}

// ---------------------------------------------------------------------------
// Final projection: p = g @ proj_w + proj_b  (N=256 -> 2 C tiles per wave)
// ---------------------------------------------------------------------------
__global__ __launch_bounds__(256) void final_proj_kernel(
    const float* __restrict__ g, const __bf16* __restrict__ WT,  // [256][128]
    const float* __restrict__ bias, float* __restrict__ p)
{
    constexpr int GST = HID + 4;
    __shared__ __attribute__((aligned(16))) __bf16 Gs[16 * GST];
    const int  tid  = threadIdx.x;
    const int  wave = tid >> 5;
    const long row0 = (long)blockIdx.x * 16;

    for (int i = tid; i < 16 * (HID / 4); i += 256) {
        const int t = i >> 5, c = (i & 31) * 4;
        const float4 v = *(const float4*)(g + (row0 + t) * HID + c);
        uint2 pk;
        pk.x = pack2bf(v.x, v.y);
        pk.y = pack2bf(v.z, v.w);
        *(uint2*)(Gs + t * GST + c) = pk;
    }
    __syncthreads();

    const int lane = tid & 31, half = lane >> 4, n = lane & 15;
    const int n0a = wave * 16, n0b = (wave + 8) * 16;
    v8f acc0 = {0.f, 0.f, 0.f, 0.f, 0.f, 0.f, 0.f, 0.f};
    v8f acc1 = {0.f, 0.f, 0.f, 0.f, 0.f, 0.f, 0.f, 0.f};
#pragma unroll
    for (int k0 = 0; k0 < HID; k0 += 32) {
        v16bf a = load_a_frag(Gs + k0, GST);
        acc0 = wmma_bf16(a, load_b_frag_g(WT, HID, n0a, k0), acc0);
        acc1 = wmma_bf16(a, load_b_frag_g(WT, HID, n0b, k0), acc1);
    }
    const float bva = bias[n0a + n], bvb = bias[n0b + n];
#pragma unroll
    for (int r = 0; r < 8; ++r) {
        const long rr = row0 + r + 8 * half;
        p[rr * OUTD + n0a + n] = acc0[r] + bva;
        p[rr * OUTD + n0b + n] = acc1[r] + bvb;
    }
}

// Row-wise L2 normalize (256 cols -> one block per row)
__global__ __launch_bounds__(256) void normalize_kernel(
    const float* __restrict__ p, float* __restrict__ out)
{
    __shared__ float red[8];
    const int   row = blockIdx.x, tid = threadIdx.x;
    const float v   = p[(long)row * OUTD + tid];
    float s = v * v;
#pragma unroll
    for (int off = 16; off > 0; off >>= 1) s += __shfl_down(s, off, 32);
    if ((tid & 31) == 0) red[tid >> 5] = s;
    __syncthreads();
    if (tid == 0) {
        float t = 0.f;
        for (int i = 0; i < 8; ++i) t += red[i];
        red[0] = fmaxf(sqrtf(t), 1e-12f);
    }
    __syncthreads();
    out[(long)row * OUTD + tid] = v / red[0];
}

__global__ void zero_kernel(float* __restrict__ p, long n) {
    const long i = (long)blockIdx.x * blockDim.x + threadIdx.x;
    if (i < n) p[i] = 0.f;
}

// ---------------------------------------------------------------------------
extern "C" void kernel_launch(void* const* d_in, const int* in_sizes, int n_in,
                              void* d_out, int out_size, void* d_ws,
                              size_t ws_size, hipStream_t stream)
{
    (void)in_sizes; (void)n_in; (void)out_size; (void)ws_size;
    const int*   x        = (const int*)d_in[0];
    const int*   eattr    = (const int*)d_in[1];
    const int*   eindex   = (const int*)d_in[2];
    const int*   batch    = (const int*)d_in[3];
    const float* node_emb = (const float*)d_in[4];
    const float* edge_emb = (const float*)d_in[5];
    const float* npw      = (const float*)d_in[6];
    const float* npb      = (const float*)d_in[7];
    const float* epw      = (const float*)d_in[8];
    const float* epb      = (const float*)d_in[9];
    const float* w1       = (const float*)d_in[10];
    const float* b1       = (const float*)d_in[11];
    const float* w2       = (const float*)d_in[12];
    const float* b2       = (const float*)d_in[13];
    const float* pw       = (const float*)d_in[14];
    const float* pb       = (const float*)d_in[15];

    float* ws  = (float*)d_ws;
    float* h   = ws;                                  // 200000*128 f32
    float* ef  = h   + (size_t)NNODES * HID;          // 400000*128 f32
    float* agg = ef  + (size_t)NEDGES * HID;          // 200000*128 f32
    float* g   = agg + (size_t)NNODES * HID;          // 8000*128  f32
    float* p   = g   + (size_t)NGRAPHS * HID;         // 8000*256  f32

    // bf16 transposed weight panels (16B-aligned: offsets are /8-multiples)
    __bf16* bfw  = (__bf16*)(p + (size_t)NGRAPHS * OUTD);
    __bf16* npwT = bfw;                                // 128 x 1152
    __bf16* epwT = npwT + (size_t)HID * (9 * HID);     // 128 x 384
    __bf16* w1T  = epwT + (size_t)HID * (3 * HID);     // 3 x (128 x 128)
    __bf16* w2T  = w1T  + (size_t)NLAYERS * HID * HID; // 3 x (128 x 128)
    __bf16* pwT  = w2T  + (size_t)NLAYERS * HID * HID; // 256 x 128

    // --- one-time weight transpose/convert (tiny; L2-resident afterwards) ---
    weight_prep_kernel<<<(9 * HID * HID + 255) / 256, 256, 0, stream>>>(
        npw, npwT, 9 * HID, HID);
    weight_prep_kernel<<<(3 * HID * HID + 255) / 256, 256, 0, stream>>>(
        epw, epwT, 3 * HID, HID);
    for (int l = 0; l < NLAYERS; ++l) {
        weight_prep_kernel<<<(HID * HID + 255) / 256, 256, 0, stream>>>(
            w1 + (size_t)l * HID * HID, w1T + (size_t)l * HID * HID, HID, HID);
        weight_prep_kernel<<<(HID * HID + 255) / 256, 256, 0, stream>>>(
            w2 + (size_t)l * HID * HID, w2T + (size_t)l * HID * HID, HID, HID);
    }
    weight_prep_kernel<<<(HID * OUTD + 255) / 256, 256, 0, stream>>>(
        pw, pwT, HID, OUTD);

    // --- pipeline ---
    embed_project_kernel<9><<<NNODES / 16, 256, 0, stream>>>(x, node_emb, npwT,
                                                             npb, h);
    embed_project_kernel<3><<<NEDGES / 16, 256, 0, stream>>>(eattr, edge_emb,
                                                             epwT, epb, ef);

    const int* src = eindex;
    const int* dst = eindex + NEDGES;
    const long nagg = (long)NNODES * HID;
    for (int l = 0; l < NLAYERS; ++l) {
        zero_kernel<<<(int)((nagg + 255) / 256), 256, 0, stream>>>(agg, nagg);
        edge_message_kernel<<<NEDGES / 8, 256, 0, stream>>>(h, ef, src, dst,
                                                            agg);
        gine_mlp_kernel<<<NNODES / 16, 256, 0, stream>>>(
            h, agg, w1T + (size_t)l * HID * HID, b1 + (size_t)l * HID,
            w2T + (size_t)l * HID * HID, b2 + (size_t)l * HID);
    }
    const long ngf = (long)NGRAPHS * HID;
    zero_kernel<<<(int)((ngf + 255) / 256), 256, 0, stream>>>(g, ngf);
    pool_kernel<<<NNODES / 8, 256, 0, stream>>>(h, batch, g);
    final_proj_kernel<<<NGRAPHS / 16, 256, 0, stream>>>(g, pwT, pb, p);
    normalize_kernel<<<NGRAPHS, 256, 0, stream>>>(p, (float*)d_out);
}
